// MultiHeadSelfAttention_55018531061801
// MI455X (gfx1250) — matrix operand order repaired and checked by reading, not compiled, not run on hardware
//
#include <hip/hip_runtime.h>
#include <hip/hip_bf16.h>

typedef __attribute__((ext_vector_type(16))) _Float16 v16h;
typedef __attribute__((ext_vector_type(8)))  float    v8f;

constexpr int BATCH  = 4;
constexpr int SEQ    = 2048;
constexpr int DMODEL = 1024;
constexpr int NH     = 16;
constexpr int DH     = 64;
constexpr float SM_SCALE  = 0.125f;              // 1/sqrt(64)
constexpr float LOG_THETA = 9.210340371976184f;  // ln(10000)

union HalfVec { v16h v; uint4 q[2]; };

__device__ __forceinline__ v8f wmma_f16(v16h a, v16h b, v8f c) {
  // D = A(16x32 f16) * B(32x16 f16) + C(16x16 f32)
  return __builtin_amdgcn_wmma_f32_16x16x32_f16(
      /*neg_a=*/false, a, /*neg_b=*/false, b,
      /*c_mod=*/(short)0, c, /*reuse_a=*/false, /*reuse_b=*/false);
}

// A-operand tile (16 rows x 32 K, f16). p -> element (row0, k0); stride in halves.
// ISA layout: lane holds row = lane&15; K chunks (lane>>4)*8..+7 and 16+(lane>>4)*8..+7.
__device__ __forceinline__ v16h load_a_tile(const _Float16* p, int stride, int lane) {
  const int row  = lane & 15;
  const int koff = (lane >> 4) << 3;  // 0 or 8
  const _Float16* q = p + row * stride + koff;
  HalfVec u;
  u.q[0] = *(const uint4*)(q);        // K = koff .. koff+7   -> VGPR 0..3
  u.q[1] = *(const uint4*)(q + 16);   // K = 16+koff .. +7    -> VGPR 4..7
  return u.v;
}

// B-operand tile (32 K x 16 cols, f16) loaded from a (N,K)-row-major array.
// ISA layout: lane holds col = lane&15; K chunks (lane>>4)*8..+7 and 16+(lane>>4)*8..+7.
__device__ __forceinline__ v16h load_b_tile(const _Float16* p, int stride, int lane) {
  const int col  = lane & 15;
  const int koff = (lane >> 4) << 3;  // 0 or 8
  const _Float16* q = p + col * stride + koff;
  HalfVec u;
  u.q[0] = *(const uint4*)(q);        // K = koff .. koff+7
  u.q[1] = *(const uint4*)(q + 16);   // K = 16+koff .. +7
  return u.v;
}

__device__ __forceinline__ float rmax16(float v) {
#pragma unroll
  for (int m = 1; m < 16; m <<= 1) v = fmaxf(v, __shfl_xor(v, m, 16));
  return v;
}
__device__ __forceinline__ float rsum16(float v) {
#pragma unroll
  for (int m = 1; m < 16; m <<= 1) v += __shfl_xor(v, m, 16);
  return v;
}

// ---------------------------------------------------------------------------
// GEMM: Y = A (M x K, f16 row-major) @ W^T, W is (N x K, f16 row-major).
// Wave tile 32x64, 4 waves/block stacked in M -> block tile 128x64.
// ---------------------------------------------------------------------------
enum { EPI_F32 = 0, EPI_QK = 1, EPI_VT = 2 };

template <int EPI>
__global__ __launch_bounds__(128) void gemm_awt_kernel(
    const _Float16* __restrict__ A, const _Float16* __restrict__ W,
    void* __restrict__ outv) {
  constexpr int N = DMODEL, K = DMODEL;
  const int wave = threadIdx.x >> 5, lane = threadIdx.x & 31;
  const int hi = lane >> 4, lo = lane & 15;
  const int m0 = (blockIdx.x * 4 + wave) * 32;
  const int n0 = blockIdx.y * 64;

  v8f c[2][4] = {};
  const _Float16* a0p = A + (size_t)m0 * K;
  const _Float16* a1p = A + (size_t)(m0 + 16) * K;

  for (int k = 0; k < K; k += 32) {
    v16h a0 = load_a_tile(a0p + k, K, lane);
    v16h a1 = load_a_tile(a1p + k, K, lane);
#pragma unroll
    for (int j = 0; j < 4; ++j) {
      v16h b = load_b_tile(W + (size_t)(n0 + 16 * j) * K + k, K, lane);
      c[0][j] = wmma_f16(a0, b, c[0][j]);
      c[1][j] = wmma_f16(a1, b, c[1][j]);
    }
  }

  // C layout: lane holds (row = r + 8*(lane>>4), col = lane&15) for VGPR r.
#pragma unroll
  for (int i = 0; i < 2; ++i)
#pragma unroll
    for (int j = 0; j < 4; ++j)
#pragma unroll
      for (int r = 0; r < 8; ++r) {
        const int m = m0 + 16 * i + r + 8 * hi;
        const int n = n0 + 16 * j + lo;
        const float val = c[i][j][r];
        if constexpr (EPI == EPI_F32) {
          ((float*)outv)[(size_t)m * N + n] = val;
        } else if constexpr (EPI == EPI_QK) {
          // [b, h, s, d] with d contiguous (A-operand layout for flash)
          const int b = m >> 11, s = m & (SEQ - 1);
          const int h = n >> 6,  d = n & (DH - 1);
          ((_Float16*)outv)[(((size_t)(b * NH + h) * SEQ) + s) * DH + d] =
              (_Float16)val;
        } else {  // EPI_VT: V transposed -> [b, h*64+d, s] with s contiguous
          ((_Float16*)outv)[((size_t)(m >> 11) * DMODEL + n) * SEQ +
                            (m & (SEQ - 1))] = (_Float16)val;
        }
      }
}

// ---------------------------------------------------------------------------
// fp32 -> fp16 conversion
// ---------------------------------------------------------------------------
__global__ void cvt_f32_f16_kernel(const float* __restrict__ src,
                                   _Float16* __restrict__ dst, int n) {
  const int i = blockIdx.x * blockDim.x + threadIdx.x;
  if (i < n) dst[i] = (_Float16)src[i];
}

// ---------------------------------------------------------------------------
// RoPE in place on f16 tensor in [b, h, s, d] layout (interleaved pairs).
// ---------------------------------------------------------------------------
__global__ void rope_kernel(_Float16* __restrict__ t,
                            const int* __restrict__ tok) {
  const int idx = blockIdx.x * blockDim.x + threadIdx.x;
  if (idx >= BATCH * NH * SEQ * (DH / 2)) return;
  const int i  = idx & 31;           // pair index 0..31
  const int s  = (idx >> 5) & (SEQ - 1);
  const int bh = idx >> 16;          // 32 * 2048 = 2^16
  const size_t base = (((size_t)bh * SEQ) + s) * DH + 2 * i;
  const float pos  = (float)tok[s];
  const float freq = __expf(-((float)(2 * i) * (1.0f / DH)) * LOG_THETA);
  float sn, cs;
  __sincosf(pos * freq, &sn, &cs);
  const float e  = (float)t[base];
  const float od = (float)t[base + 1];
  t[base]     = (_Float16)(e * cs - od * sn);
  t[base + 1] = (_Float16)(e * sn + od * cs);
}

// ---------------------------------------------------------------------------
// Flash attention: one wave per 16-query tile, 32 keys per iteration.
// Q,K in [b,h,s,d] (f16), V in [b,h,d,s] (f16). Output [b,s,h,d] (f16).
// ---------------------------------------------------------------------------
__global__ __launch_bounds__(256) void flash_attn_kernel(
    const _Float16* __restrict__ Qh, const _Float16* __restrict__ Kh,
    const _Float16* __restrict__ Vt, _Float16* __restrict__ Oh) {
  __shared__ _Float16 lds[8 * 16 * 32];  // 1KB per wave (16x32 P tile)
  const int wave = threadIdx.x >> 5, lane = threadIdx.x & 31;
  const int hi = lane >> 4, lo = lane & 15;
  const int bh = blockIdx.y, b = bh >> 4, h = bh & 15;
  const int q0 = blockIdx.x * 128 + wave * 16;

  const _Float16* Qbase = Qh + ((size_t)bh * SEQ + q0) * DH;
  const v16h qa0 = load_a_tile(Qbase,      DH, lane);  // dh 0..31
  const v16h qa1 = load_a_tile(Qbase + 32, DH, lane);  // dh 32..63

  v8f o[4] = {};
  float mr[8], lr[8];
#pragma unroll
  for (int r = 0; r < 8; ++r) { mr[r] = -3.0e38f; lr[r] = 0.0f; }

  _Float16* my = lds + wave * (16 * 32);
  const _Float16* Kbh = Kh + (size_t)bh * SEQ * DH;
  const _Float16* Vbh = Vt + ((size_t)b * DMODEL + h * DH) * SEQ;

  for (int kb = 0; kb < q0 + 16; kb += 32) {
    const _Float16* Kp = Kbh + (size_t)kb * DH;
    const v16h k0a = load_b_tile(Kp,                DH, lane);  // keys kb..+15, dh 0..31
    const v16h k0b = load_b_tile(Kp + 32,           DH, lane);  // keys kb..+15, dh 32..63
    const v16h k1a = load_b_tile(Kp + 16 * DH,      DH, lane);  // keys kb+16..+31
    const v16h k1b = load_b_tile(Kp + 16 * DH + 32, DH, lane);

    v8f s0 = {}, s1 = {};
    s0 = wmma_f16(qa0, k0a, s0); s0 = wmma_f16(qa1, k0b, s0);
    s1 = wmma_f16(qa0, k1a, s1); s1 = wmma_f16(qa1, k1b, s1);

    const bool need_mask = (kb + 32 > q0);  // wave-uniform
    float p0[8], p1[8];
#pragma unroll
    for (int r = 0; r < 8; ++r) {
      float x0 = s0[r] * SM_SCALE;
      float x1 = s1[r] * SM_SCALE;
      if (need_mask) {  // branch-free per-lane select: keeps EXEC all-ones
        const int q = q0 + r + 8 * hi;
        x0 = (kb + lo      > q) ? -3.0e38f : x0;
        x1 = (kb + 16 + lo > q) ? -3.0e38f : x1;
      }
      const float rm   = rmax16(fmaxf(x0, x1));
      const float mnew = fmaxf(mr[r], rm);
      const float al   = __expf(mr[r] - mnew);
      mr[r] = mnew;
      p0[r] = __expf(x0 - mnew);
      p1[r] = __expf(x1 - mnew);
      lr[r] = lr[r] * al + rsum16(p0[r] + p1[r]);
#pragma unroll
      for (int cc = 0; cc < 4; ++cc) o[cc][r] *= al;
    }

    // P (C layout) -> f16 A-operand layout via per-wave LDS round-trip.
#pragma unroll
    for (int r = 0; r < 8; ++r) {
      const int m = (r + 8 * hi) * 32;
      my[m + lo]      = (_Float16)p0[r];
      my[m + 16 + lo] = (_Float16)p1[r];
    }
    asm volatile("s_wait_dscnt 0" ::: "memory");  // CDNA5 DS counter fence
    HalfVec up;
    up.q[0] = *(const uint4*)(my + lo * 32 + hi * 8);
    up.q[1] = *(const uint4*)(my + lo * 32 + 16 + hi * 8);
    const v16h pa = up.v;

#pragma unroll
    for (int cc = 0; cc < 4; ++cc) {
      const v16h vb = load_b_tile(Vbh + (size_t)(16 * cc) * SEQ + kb, SEQ, lane);
      o[cc] = wmma_f16(pa, vb, o[cc]);
    }
  }

  float inv[8];
#pragma unroll
  for (int r = 0; r < 8; ++r) inv[r] = 1.0f / lr[r];
#pragma unroll
  for (int cc = 0; cc < 4; ++cc)
#pragma unroll
    for (int r = 0; r < 8; ++r) {
      const size_t row = (size_t)b * SEQ + q0 + r + 8 * hi;
      Oh[row * DMODEL + h * DH + 16 * cc + lo] =
          (_Float16)(o[cc][r] * inv[r]);
    }
}

// ---------------------------------------------------------------------------
extern "C" void kernel_launch(void* const* d_in, const int* in_sizes, int n_in,
                              void* d_out, int out_size, void* d_ws,
                              size_t ws_size, hipStream_t stream) {
  (void)in_sizes; (void)n_in; (void)out_size; (void)ws_size;
  const float* x  = (const float*)d_in[0];
  const float* WQ = (const float*)d_in[1];
  const float* WK = (const float*)d_in[2];
  const float* WV = (const float*)d_in[3];
  const float* WO = (const float*)d_in[4];
  const int*  tok = (const int*)d_in[5];

  constexpr int XN = BATCH * SEQ * DMODEL;  // 8388608
  constexpr int WN = DMODEL * DMODEL;       // 1048576

  _Float16* p   = (_Float16*)d_ws;          // total ~92 MB of f16 scratch
  _Float16* xh  = p; p += XN;
  _Float16* wqh = p; p += WN;
  _Float16* wkh = p; p += WN;
  _Float16* wvh = p; p += WN;
  _Float16* woh = p; p += WN;
  _Float16* qh  = p; p += XN;
  _Float16* kh  = p; p += XN;
  _Float16* vt  = p; p += XN;
  _Float16* ah  = p; p += XN;

  cvt_f32_f16_kernel<<<(XN + 255) / 256, 256, 0, stream>>>(x,  xh,  XN);
  cvt_f32_f16_kernel<<<(WN + 255) / 256, 256, 0, stream>>>(WQ, wqh, WN);
  cvt_f32_f16_kernel<<<(WN + 255) / 256, 256, 0, stream>>>(WK, wkh, WN);
  cvt_f32_f16_kernel<<<(WN + 255) / 256, 256, 0, stream>>>(WV, wvh, WN);
  cvt_f32_f16_kernel<<<(WN + 255) / 256, 256, 0, stream>>>(WO, woh, WN);

  const dim3 ggrid(BATCH * SEQ / 128, DMODEL / 64);  // (64, 16)
  gemm_awt_kernel<EPI_QK><<<ggrid, 128, 0, stream>>>(xh, wqh, qh);
  gemm_awt_kernel<EPI_QK><<<ggrid, 128, 0, stream>>>(xh, wkh, kh);
  gemm_awt_kernel<EPI_VT><<<ggrid, 128, 0, stream>>>(xh, wvh, vt);

  const int nrope = BATCH * NH * SEQ * (DH / 2);
  rope_kernel<<<(nrope + 255) / 256, 256, 0, stream>>>(qh, tok);
  rope_kernel<<<(nrope + 255) / 256, 256, 0, stream>>>(kh, tok);

  flash_attn_kernel<<<dim3(SEQ / 128, BATCH * NH), 256, 0, stream>>>(qh, kh, vt, ah);

  gemm_awt_kernel<EPI_F32><<<ggrid, 128, 0, stream>>>(ah, woh, (float*)d_out);
}
